// MAB_17471926960685
// MI455X (gfx1250) — matrix operand order repaired and checked by reading, not compiled, not run on hardware
//
#include <hip/hip_runtime.h>
#include <hip/hip_bf16.h>
#include <math.h>

// ---------------------------------------------------------------------------
// MAB block for MI455X (gfx1250, wave32, WMMA).
// All matmuls use v_wmma_f32_16x16x32_bf16 (bf16 in, fp32 accumulate).
// Attention is fused flash-style: 256MB score matrix never materialized;
// add_enc/mult_enc (268MB fp32 = the HBM floor, > 192MB L2) are streamed with
// non-temporal loads so they don't evict the K/V/Q/Hid working set from L2.
// ---------------------------------------------------------------------------

typedef __attribute__((ext_vector_type(16))) __bf16 v16bf;
typedef __attribute__((ext_vector_type(8)))  float  v8f;

#define WMMA_BF16(a, b, c) \
  __builtin_amdgcn_wmma_f32_16x16x32_bf16(false, (a), false, (b), (short)0, (c), false, false)

// Problem constants
#define NB    2      // batch
#define NTOK  2048   // sequence length
#define DEMB  256    // embed dim
#define NH    8      // heads
#define DSH   32     // dim per head-split
#define MROWS 4096   // NB*NTOK
#define DFF   1024   // 4*DO

static __device__ __forceinline__ __bf16 to_bf16(float f) {
  union { float f; unsigned u; } in; in.f = f;
  unsigned r = in.u + 0x7FFFu + ((in.u >> 16) & 1u);   // round-to-nearest-even
  union { unsigned short s; __bf16 h; } o; o.s = (unsigned short)(r >> 16);
  return o.h;
}

union FragU { v16bf v; uint4 q[2]; };

// A-matrix fragment (16xK=32, bf16). Per CDNA5 ISA 7.12.2: lane%16 = row M,
// lane/16 selects K-halves: elems 0..7 -> K=8*half+e, elems 8..15 -> K=16+8*half+e.
static __device__ __forceinline__ v16bf load_a_frag(const __bf16* base, int row0,
                                                    int ld, int k0, int lane) {
  const int colh = lane & 15, half = lane >> 4;
  const __bf16* p = base + (size_t)(row0 + colh) * ld + k0;
  FragU f;
  f.q[0] = *reinterpret_cast<const uint4*>(p + 8 * half);
  f.q[1] = *reinterpret_cast<const uint4*>(p + 16 + 8 * half);
  return f.v;
}

// B-matrix fragment (K=32 x 16, bf16) from a row-major (N x K) source (i.e. W
// for X@W^T). lane%16 = column N, elems 0..7 -> K=8*half+e, 8..15 -> K=16+8*half+e.
static __device__ __forceinline__ v16bf load_b_frag(const __bf16* base, int n0,
                                                    int ld, int k0, int lane) {
  const int colh = lane & 15, half = lane >> 4;
  const __bf16* p = base + (size_t)(n0 + colh) * ld + k0;
  FragU f;
  f.q[0] = *reinterpret_cast<const uint4*>(p + 8 * half);
  f.q[1] = *reinterpret_cast<const uint4*>(p + 16 + 8 * half);
  return f.v;
}

static __device__ __forceinline__ float gelu_new(float x) {
  float x3 = x * x * x;
  return 0.5f * x * (1.0f + tanhf(0.7978845608028654f * (x + 0.044715f * x3)));
}

// ---------------------------------------------------------------------------
// fp32 -> bf16 elementwise convert
__global__ void mab_cvt(const float* __restrict__ in, __bf16* __restrict__ out, int n) {
  int i = blockIdx.x * blockDim.x + threadIdx.x;
  if (i < n) out[i] = to_bf16(in[i]);
}

// ---------------------------------------------------------------------------
// LayerNorm over rows of 256, output bf16. One wave per row.
__global__ void mab_layernorm(const float* __restrict__ x, const float* __restrict__ g,
                              const float* __restrict__ bt, __bf16* __restrict__ o,
                              int mrows) {
  const int lane = threadIdx.x & 31, wave = threadIdx.x >> 5;
  const int row = blockIdx.x * 8 + wave;
  if (row >= mrows) return;
  const float* xr = x + (size_t)row * DEMB;
  float v[8]; float s = 0.f;
#pragma unroll
  for (int i = 0; i < 8; i++) { v[i] = xr[i * 32 + lane]; s += v[i]; }
#pragma unroll
  for (int off = 16; off; off >>= 1) s += __shfl_xor(s, off, 32);
  const float mu = s * (1.f / 256.f);
  float q = 0.f;
#pragma unroll
  for (int i = 0; i < 8; i++) { float d = v[i] - mu; q += d * d; }
#pragma unroll
  for (int off = 16; off; off >>= 1) q += __shfl_xor(q, off, 32);
  const float rinv = rsqrtf(q * (1.f / 256.f) + 1e-5f);
#pragma unroll
  for (int i = 0; i < 8; i++) {
    int c = i * 32 + lane;
    o[(size_t)row * DEMB + c] = to_bf16((v[i] - mu) * rinv * g[c] + bt[c]);
  }
}

// ---------------------------------------------------------------------------
// Generic bf16 GEMM  C = A(MxK) @ Bw(NxK)^T  with fused epilogues.
// One wave computes a 32x32 tile: 4 WMMAs per 32-wide K step.
// mode 0: +bias -> Qf fp32 (M,256)  AND  Qh bf16 split-heads (H*B,N,ds)
// mode 1: +bias -> Kh bf16 split-heads
// mode 2: +bias -> Vt bf16 TRANSPOSED split-heads (H*B,ds,N)  (for P@V B-frags)
// mode 3: +bias + res -> f32o (mix + residual)
// mode 4: +res        -> f32o (ffn out + residual)
__global__ void mab_gemm(const __bf16* __restrict__ A, const __bf16* __restrict__ Bw,
                         const float* __restrict__ bias, int M, int N, int K, int mode,
                         float* __restrict__ f32o, __bf16* __restrict__ bfo,
                         const float* __restrict__ res) {
  const int lane = threadIdx.x & 31;
  const int gw = (int)((blockIdx.x * blockDim.x + threadIdx.x) >> 5);
  const int tn = N >> 5;
  if (gw >= (M >> 5) * tn) return;
  const int m0 = (gw / tn) * 32, n0 = (gw % tn) * 32;

  v8f z = {0.f, 0.f, 0.f, 0.f, 0.f, 0.f, 0.f, 0.f};
  v8f acc00 = z, acc01 = z, acc10 = z, acc11 = z;
  for (int k0 = 0; k0 < K; k0 += 32) {
    v16bf a0 = load_a_frag(A, m0,      K, k0, lane);
    v16bf a1 = load_a_frag(A, m0 + 16, K, k0, lane);
    v16bf b0 = load_b_frag(Bw, n0,      K, k0, lane);
    v16bf b1 = load_b_frag(Bw, n0 + 16, K, k0, lane);
    acc00 = WMMA_BF16(a0, b0, acc00);
    acc01 = WMMA_BF16(a0, b1, acc01);
    acc10 = WMMA_BF16(a1, b0, acc10);
    acc11 = WMMA_BF16(a1, b1, acc11);
  }

  const int colh = lane & 15, half = lane >> 4;
  v8f accs[4] = {acc00, acc01, acc10, acc11};
#pragma unroll
  for (int t = 0; t < 4; t++) {
    const int i = t >> 1, j = t & 1;
    const int n = n0 + 16 * j + colh;
    const float bv = bias ? bias[n] : 0.f;
#pragma unroll
    for (int r = 0; r < 8; r++) {
      const int m = m0 + 16 * i + 8 * half + r;   // C layout: M = r + 8*(lane>=16)
      float v = accs[t][r] + bv;
      if (mode == 0) {
        f32o[(size_t)m * N + n] = v;
        int hh = n >> 5, dd = n & 31, bb = m >> 11, tok = m & (NTOK - 1);
        bfo[(((size_t)(hh * NB + bb) * NTOK + tok) << 5) + dd] = to_bf16(v);
      } else if (mode == 1) {
        int hh = n >> 5, dd = n & 31, bb = m >> 11, tok = m & (NTOK - 1);
        bfo[(((size_t)(hh * NB + bb) * NTOK + tok) << 5) + dd] = to_bf16(v);
      } else if (mode == 2) {
        int hh = n >> 5, dd = n & 31, bb = m >> 11, tok = m & (NTOK - 1);
        bfo[((size_t)(hh * NB + bb) * DSH + dd) * NTOK + tok] = to_bf16(v);
      } else {
        f32o[(size_t)m * N + n] = v + res[(size_t)m * N + n];
      }
    }
  }
}

// ---------------------------------------------------------------------------
// Gated FFN first stage: G = gelu_new(Hr@wi0^T) * (Hr@wi1^T), bf16 out.
// 8 WMMAs per 32-K step (two B matrices share the A fragments).
__global__ void mab_ffn1(const __bf16* __restrict__ A, const __bf16* __restrict__ W0,
                         const __bf16* __restrict__ W1, __bf16* __restrict__ G,
                         int M, int N, int K) {
  const int lane = threadIdx.x & 31;
  const int gw = (int)((blockIdx.x * blockDim.x + threadIdx.x) >> 5);
  const int tn = N >> 5;
  if (gw >= (M >> 5) * tn) return;
  const int m0 = (gw / tn) * 32, n0 = (gw % tn) * 32;

  v8f z = {0.f, 0.f, 0.f, 0.f, 0.f, 0.f, 0.f, 0.f};
  v8f u0[4] = {z, z, z, z}, u1[4] = {z, z, z, z};
  for (int k0 = 0; k0 < K; k0 += 32) {
    v16bf a0 = load_a_frag(A, m0,      K, k0, lane);
    v16bf a1 = load_a_frag(A, m0 + 16, K, k0, lane);
    v16bf b00 = load_b_frag(W0, n0,      K, k0, lane);
    v16bf b01 = load_b_frag(W0, n0 + 16, K, k0, lane);
    v16bf b10 = load_b_frag(W1, n0,      K, k0, lane);
    v16bf b11 = load_b_frag(W1, n0 + 16, K, k0, lane);
    u0[0] = WMMA_BF16(a0, b00, u0[0]);
    u0[1] = WMMA_BF16(a0, b01, u0[1]);
    u0[2] = WMMA_BF16(a1, b00, u0[2]);
    u0[3] = WMMA_BF16(a1, b01, u0[3]);
    u1[0] = WMMA_BF16(a0, b10, u1[0]);
    u1[1] = WMMA_BF16(a0, b11, u1[1]);
    u1[2] = WMMA_BF16(a1, b10, u1[2]);
    u1[3] = WMMA_BF16(a1, b11, u1[3]);
  }
  const int colh = lane & 15, half = lane >> 4;
#pragma unroll
  for (int t = 0; t < 4; t++) {
    const int i = t >> 1, j = t & 1;
    const int n = n0 + 16 * j + colh;
#pragma unroll
    for (int r = 0; r < 8; r++) {
      const int m = m0 + 16 * i + 8 * half + r;
      G[(size_t)m * N + n] = to_bf16(gelu_new(u0[t][r]) * u1[t][r]);
    }
  }
}

// ---------------------------------------------------------------------------
// Fused flash attention with additive + multiplicative encodings.
// grid = (NTOK/128, H*B), block = 256 (8 waves). Each wave owns a 16-row q
// tile, streams Nkv in chunks of 32, online softmax, mult_enc folded into the
// P@V numerator, denominator tracked separately (mult applies post-softmax).
// Encoding streams use non-temporal loads (268MB read-once > 192MB L2).
__global__ void mab_attention(const __bf16* __restrict__ Qh, const __bf16* __restrict__ Kh,
                              const __bf16* __restrict__ Vt,
                              const float* __restrict__ add_enc,
                              const float* __restrict__ mult_enc,
                              __bf16* __restrict__ MHc) {
  __shared__ __align__(16) __bf16 Pm[8][16][32];   // per-wave 16x32 P tile
  const int lane = threadIdx.x & 31, wave = threadIdx.x >> 5;
  const int colh = lane & 15, half = lane >> 4;
  const int hb = blockIdx.y;            // flat head-batch index i = h*B + b
  const int h = hb >> 1, b = hb & 1;
  const int eidx = hb & (NH - 1);       // torch tile quirk: enc head = i % H
  const int q0 = (blockIdx.x * 8 + wave) * 16;

  const __bf16* Qp = Qh + (size_t)hb * NTOK * DSH;
  const __bf16* Kp = Kh + (size_t)hb * NTOK * DSH;
  const __bf16* Vp = Vt + (size_t)hb * DSH * NTOK;
  const float*  Ap = add_enc  + (size_t)eidx * NTOK * NTOK;
  const float*  Xp = mult_enc + (size_t)eidx * NTOK * NTOK;

  const v16bf qa = load_a_frag(Qp, q0, DSH, 0, lane);  // whole ds=32 in one frag
  v8f z = {0.f, 0.f, 0.f, 0.f, 0.f, 0.f, 0.f, 0.f};
  v8f acc0 = z, acc1 = z;                              // output cols 0-15 / 16-31
  float mrun[8], drun[8];
#pragma unroll
  for (int r = 0; r < 8; r++) { mrun[r] = -1e30f; drun[r] = 0.f; }
  const float scale = 0.0625f;                         // 1/sqrt(256)
  const int qrb = q0 + 8 * half;                       // C-layout row base

  for (int k0 = 0; k0 < NTOK; k0 += 32) {
    // scores: one WMMA per 16x16 tile (K = ds = 32)
    v16bf kb0 = load_b_frag(Kp, k0,      DSH, 0, lane);
    v16bf kb1 = load_b_frag(Kp, k0 + 16, DSH, 0, lane);
    v8f s0 = WMMA_BF16(qa, kb0, z);
    v8f s1 = WMMA_BF16(qa, kb1, z);

    const int c0 = k0 + colh, c1 = k0 + 16 + colh;
#pragma unroll
    for (int r = 0; r < 8; r++) {
      const size_t ro = (size_t)(qrb + r) * NTOK;
      // streamed-once encodings: non-temporal (don't thrash L2)
      float a0v = __builtin_nontemporal_load(Ap + ro + c0);
      float a1v = __builtin_nontemporal_load(Ap + ro + c1);
      float x0v = __builtin_nontemporal_load(Xp + ro + c0);
      float x1v = __builtin_nontemporal_load(Xp + ro + c1);
      float t0 = (s0[r] + a0v) * scale;
      float t1 = (s1[r] + a1v) * scale;
      float tm = fmaxf(t0, t1);
#pragma unroll
      for (int off = 8; off; off >>= 1) tm = fmaxf(tm, __shfl_xor(tm, off, 32));
      float nm = fmaxf(mrun[r], tm);
      float corr = __expf(mrun[r] - nm);
      mrun[r] = nm;
      float p0 = __expf(t0 - nm);
      float p1 = __expf(t1 - nm);
      float rs = p0 + p1;
#pragma unroll
      for (int off = 8; off; off >>= 1) rs += __shfl_xor(rs, off, 32);
      drun[r] = drun[r] * corr + rs;
      acc0[r] *= corr;
      acc1[r] *= corr;
      // numerator carries mult_enc; C-layout -> row-major LDS tile
      Pm[wave][8 * half + r][colh]      = to_bf16(p0 * x0v);
      Pm[wave][8 * half + r][16 + colh] = to_bf16(p1 * x1v);
    }
    __syncthreads();
    // re-read P as an A-layout fragment (transpose via LDS)
    FragU pf;
    const __bf16* pp = &Pm[wave][colh][0];
    pf.q[0] = *reinterpret_cast<const uint4*>(pp + 8 * half);
    pf.q[1] = *reinterpret_cast<const uint4*>(pp + 16 + 8 * half);
    // V stored transposed (ds x N) so B fragments are contiguous in k
    v16bf vb0 = load_b_frag(Vp, 0,  NTOK, k0, lane);
    v16bf vb1 = load_b_frag(Vp, 16, NTOK, k0, lane);
    acc0 = WMMA_BF16(pf.v, vb0, acc0);
    acc1 = WMMA_BF16(pf.v, vb1, acc1);
    __syncthreads();
  }

  // normalize and write merged heads: (b, tok, h, d) == (B,N,DE) layout
#pragma unroll
  for (int r = 0; r < 8; r++) {
    const float inv = 1.f / drun[r];
    const int tok = qrb + r;
    size_t o = (((size_t)(b * NTOK + tok) * NH) + h) * DSH;
    MHc[o + colh]      = to_bf16(acc0[r] * inv);
    MHc[o + 16 + colh] = to_bf16(acc1[r] * inv);
  }
}

// ---------------------------------------------------------------------------
extern "C" void kernel_launch(void* const* d_in, const int* in_sizes, int n_in,
                              void* d_out, int out_size, void* d_ws, size_t ws_size,
                              hipStream_t stream) {
  const float* X        = (const float*)d_in[0];
  const float* Y        = (const float*)d_in[1];
  const float* Wq       = (const float*)d_in[2];
  const float* bq       = (const float*)d_in[3];
  const float* Wk       = (const float*)d_in[4];
  const float* bk       = (const float*)d_in[5];
  const float* Wv       = (const float*)d_in[6];
  const float* bv       = (const float*)d_in[7];
  const float* Wmix     = (const float*)d_in[8];
  const float* bmix     = (const float*)d_in[9];
  const float* g0       = (const float*)d_in[10];
  const float* b0       = (const float*)d_in[11];
  const float* g1       = (const float*)d_in[12];
  const float* b1       = (const float*)d_in[13];
  const float* add_enc  = (const float*)d_in[14];
  const float* mult_enc = (const float*)d_in[15];
  const float* wi0      = (const float*)d_in[16];
  const float* wi1      = (const float*)d_in[17];
  const float* wo       = (const float*)d_in[18];
  float* out = (float*)d_out;

  // workspace carve-up (~32.5 MB total)
  char* w = (char*)d_ws;
  auto take = [&](size_t bytes) -> char* {
    char* p = w; w += (bytes + 255) & ~(size_t)255; return p;
  };
  __bf16* Wqc   = (__bf16*)take((size_t)DEMB * DEMB * 2);
  __bf16* Wkc   = (__bf16*)take((size_t)DEMB * DEMB * 2);
  __bf16* Wvc   = (__bf16*)take((size_t)DEMB * DEMB * 2);
  __bf16* Wmixc = (__bf16*)take((size_t)DEMB * DEMB * 2);
  __bf16* wi0c  = (__bf16*)take((size_t)DFF * DEMB * 2);
  __bf16* wi1c  = (__bf16*)take((size_t)DFF * DEMB * 2);
  __bf16* woc   = (__bf16*)take((size_t)DEMB * DFF * 2);
  __bf16* Yc    = (__bf16*)take((size_t)MROWS * DEMB * 2);
  __bf16* Xn    = (__bf16*)take((size_t)MROWS * DEMB * 2);
  float*  Qf    = (float*)take((size_t)MROWS * DEMB * 4);
  __bf16* Qh    = (__bf16*)take((size_t)MROWS * DEMB * 2);
  __bf16* Kh    = (__bf16*)take((size_t)MROWS * DEMB * 2);
  __bf16* Vt    = (__bf16*)take((size_t)MROWS * DEMB * 2);
  __bf16* MHc   = (__bf16*)take((size_t)MROWS * DEMB * 2);
  float*  Hid   = (float*)take((size_t)MROWS * DEMB * 4);
  __bf16* Hr    = (__bf16*)take((size_t)MROWS * DEMB * 2);
  __bf16* Gc    = (__bf16*)take((size_t)MROWS * DFF * 2);

  auto cvt = [&](const float* s, __bf16* d, int n) {
    mab_cvt<<<(n + 255) / 256, 256, 0, stream>>>(s, d, n);
  };
  auto gemm = [&](const __bf16* A, const __bf16* Bw, const float* bias,
                  int M, int N, int K, int mode, float* fo, __bf16* bo,
                  const float* res) {
    int waves = (M >> 5) * (N >> 5);
    mab_gemm<<<(waves * 32 + 255) / 256, 256, 0, stream>>>(A, Bw, bias, M, N, K,
                                                           mode, fo, bo, res);
  };

  // weight / input conversions to bf16
  cvt(Wq,   Wqc,   DEMB * DEMB);
  cvt(Wk,   Wkc,   DEMB * DEMB);
  cvt(Wv,   Wvc,   DEMB * DEMB);
  cvt(Wmix, Wmixc, DEMB * DEMB);
  cvt(wi0,  wi0c,  DFF * DEMB);
  cvt(wi1,  wi1c,  DFF * DEMB);
  cvt(wo,   woc,   DEMB * DFF);
  cvt(Y,    Yc,    MROWS * DEMB);

  // pre-LN on X
  mab_layernorm<<<MROWS / 8, 256, 0, stream>>>(X, g0, b0, Xn, MROWS);

  // projections (Q also kept fp32 as the residual)
  gemm(Xn, Wqc, bq, MROWS, DEMB, DEMB, 0, Qf, Qh, nullptr);
  gemm(Yc, Wkc, bk, MROWS, DEMB, DEMB, 1, nullptr, Kh, nullptr);
  gemm(Yc, Wvc, bv, MROWS, DEMB, DEMB, 2, nullptr, Vt, nullptr);

  // fused flash attention with encodings
  mab_attention<<<dim3(NTOK / 128, NH * NB), 256, 0, stream>>>(
      Qh, Kh, Vt, add_enc, mult_enc, MHc);

  // mix + bias + Q residual
  gemm(MHc, Wmixc, bmix, MROWS, DEMB, DEMB, 3, Hid, nullptr, Qf);

  // second LN, gated FFN, final residual
  mab_layernorm<<<MROWS / 8, 256, 0, stream>>>(Hid, g1, b1, Hr, MROWS);
  {
    int waves = (MROWS >> 5) * (DFF >> 5);
    mab_ffn1<<<(waves * 32 + 255) / 256, 256, 0, stream>>>(Hr, wi0c, wi1c, Gc,
                                                           MROWS, DFF, DEMB);
  }
  gemm(Gc, woc, nullptr, MROWS, DEMB, DFF, 4, out, nullptr, Hid);
}
